// DGCNN_32014686225100
// MI455X (gfx1250) — compile-verified, matrix operand order checked by reading, not run on hardware
//
#include <hip/hip_runtime.h>
#include <hip/hip_bf16.h>
#include <float.h>

// ---------------------------------------------------------------------------
// DGCNN feature extractor for MI455X (gfx1250, wave32, WMMA).
// All GEMMs (knn gram, edge conv, skip conv) use v_wmma_f32_16x16x32_bf16
// with f32 accumulation. Activations live as (N, 64) bf16 rows (zero-padded
// channels), so gram/edge-conv (C=64) operand fetches are unconditional b128 chunks.
// ---------------------------------------------------------------------------

typedef __attribute__((ext_vector_type(16))) __bf16 v16bf;
typedef __attribute__((ext_vector_type(8)))  __bf16 v8bf;
typedef __attribute__((ext_vector_type(8)))  float  v8f;

#define NBATCH 8
#define NPT    2048
#define KNB    20
#define NK     (NPT * KNB)       // 40960 columns per batch
#define CMAXP  64                // channel pitch (bf16 row length)

// K held by element e of a 16-bit 16x32 A/B operand, lane-half `half`
// (ISA 7.12.2): e<8 -> c0+8*half+e ; e>=8 -> c0+16+8*half+(e-8).
__device__ __forceinline__ int wmma_kmap(int e, int half) {
    int v = e >> 1, p = e & 1;
    int base = (v < 4) ? (2 * v + p) : (16 + 2 * (v - 4) + p);
    return base + 8 * half;
}

__device__ __forceinline__ v8bf ldv8(const __bf16* p) {
    return *(const v8bf*)p;
}
__device__ __forceinline__ v16bf cat16(v8bf lo, v8bf hi) {
    return __builtin_shufflevector(lo, hi, 0, 1, 2, 3, 4, 5, 6, 7,
                                            8, 9, 10, 11, 12, 13, 14, 15);
}

// ---------------------------------------------------------------------------
// Prep: transpose input points (B,N,3) -> channel-major (B,3,N) f32
// ---------------------------------------------------------------------------
__global__ void transpose_in_kernel(const float* __restrict__ x,
                                    float* __restrict__ xa) {
    int t = blockIdx.x * blockDim.x + threadIdx.x;
    if (t >= NBATCH * NPT) return;
    int b = t / NPT, n = t % NPT;
    float* o = xa + (size_t)b * CMAXP * NPT;
#pragma unroll
    for (int c = 0; c < 3; ++c)
        o[(size_t)c * NPT + n] = x[((size_t)b * NPT + n) * 3 + c];
}

// Per-layer: squared norms + zero-padded bf16 rows (B, N, CMAXP)
__global__ void prep_kernel(const float* __restrict__ xin,   // (B,C,N) f32
                            __bf16* __restrict__ xbf,        // (B,N,64)
                            float* __restrict__ xx, int C) {
    int t = blockIdx.x * blockDim.x + threadIdx.x;
    if (t >= NBATCH * NPT) return;
    int b = t / NPT, n = t % NPT;
    const float* xi  = xin + (size_t)b * CMAXP * NPT;
    __bf16*      row = xbf + ((size_t)b * NPT + n) * CMAXP;
    float s = 0.f;
    for (int c = 0; c < CMAXP; ++c) {
        float v = (c < C) ? xi[(size_t)c * NPT + n] : 0.f;
        s += v * v;
        row[c] = (__bf16)v;
    }
    xx[t] = s;
}

// Weight cast with zero padding: (rows,csrc) f32 -> (rows,cdst) bf16
__global__ void wcast_pad_kernel(const float* __restrict__ w,
                                 __bf16* __restrict__ o,
                                 int rows, int csrc, int cdst) {
    int t = blockIdx.x * blockDim.x + threadIdx.x;
    if (t >= rows * cdst) return;
    int r = t / cdst, c = t % cdst;
    o[t] = (c < csrc) ? (__bf16)w[r * csrc + c] : (__bf16)0.0f;
}

// ---------------------------------------------------------------------------
// Gram / pairwise distance: pd[n,m] = 2*<x_n,x_m> - |x_n|^2 - |x_m|^2
// One wave computes a 16(n) x 64(m) strip: A loaded once per K-step,
// 4 B tiles, 4 WMMAs. All operand fetches are b128 chunks, no guards.
// ---------------------------------------------------------------------------
__global__ void gram_pd_kernel(const __bf16* __restrict__ xbf, // (N,64) slice
                               const float*  __restrict__ xx,  // (N) slice
                               float* __restrict__ pd,         // (N,N)
                               int Cpad) {                     // 32 or 64
    int lane = threadIdx.x;
    int half = lane >> 4, l15 = lane & 15;
    int n0 = blockIdx.x * 16;
    int m0 = blockIdx.y * 64;
    const __bf16* arow = xbf + (size_t)(n0 + l15) * CMAXP;
    v8f acc[4] = {};
    for (int c0 = 0; c0 < Cpad; c0 += 32) {
        int klo = c0 + 8 * half;
        int khi = klo + 16;
        v16bf a = cat16(ldv8(arow + klo), ldv8(arow + khi));
#pragma unroll
        for (int tI = 0; tI < 4; ++tI) {
            const __bf16* brow = xbf + (size_t)(m0 + 16 * tI + l15) * CMAXP;
            v16bf bm = cat16(ldv8(brow + klo), ldv8(brow + khi));
            acc[tI] = __builtin_amdgcn_wmma_f32_16x16x32_bf16(
                false, a, false, bm, (short)0, acc[tI], false, false);
        }
    }
#pragma unroll
    for (int tI = 0; tI < 4; ++tI) {
        int col = m0 + 16 * tI + l15;
        float xc = xx[col];
#pragma unroll
        for (int j = 0; j < 8; ++j) {
            int row = n0 + j + 8 * half;
            pd[(size_t)row * NPT + col] = 2.0f * acc[tI][j] - xx[row] - xc;
        }
    }
}

// ---------------------------------------------------------------------------
// Top-20 per row: float4 candidate loads + register insertion sort.
// Strict '>' keeps lower indices first on ties (matches jax.lax.top_k).
// ---------------------------------------------------------------------------
__global__ void topk_kernel(const float* __restrict__ pd,
                            int* __restrict__ idx) {        // (N,KNB) slice
    int n = blockIdx.x * blockDim.x + threadIdx.x;
    if (n >= NPT) return;
    float best[KNB]; int bi[KNB];
#pragma unroll
    for (int j = 0; j < KNB; ++j) { best[j] = -3.0e38f; bi[j] = 0; }
    const float4* row = (const float4*)(pd + (size_t)n * NPT);
    for (int m4 = 0; m4 < NPT / 4; ++m4) {
        float4 q = row[m4];
        float dv[4] = {q.x, q.y, q.z, q.w};
#pragma unroll
        for (int s = 0; s < 4; ++s) {
            float d = dv[s];
            if (d > best[KNB - 1]) {
                best[KNB - 1] = d; bi[KNB - 1] = m4 * 4 + s;
#pragma unroll
                for (int j = KNB - 1; j > 0; --j) {
                    if (best[j] > best[j - 1]) {
                        float tf = best[j]; best[j] = best[j - 1]; best[j - 1] = tf;
                        int   ti = bi[j];   bi[j]   = bi[j - 1];   bi[j - 1]   = ti;
                    }
                }
            }
        }
    }
#pragma unroll
    for (int j = 0; j < KNB; ++j) idx[n * KNB + j] = bi[j];
}

// ---------------------------------------------------------------------------
// Edge conv: y[o,(n,k)] = sum_c W[o,c] * feat[c,(n,k)],
// feat = [x_nbr - x_ctr ; x_ctr] gathered from bf16 rows.
// Fast chunked path for C=64; tiny branch-light path for C=3 (layer 1,
// where zero padding makes nbr-ctr self-zeroing past channel 2).
// ---------------------------------------------------------------------------
__global__ void edgeconv_kernel(const __bf16* __restrict__ Wbf, // (64,twoCpad)
                                const __bf16* __restrict__ xbf, // (B,N,64)
                                const int*    __restrict__ idx, // (B,N,KNB)
                                float* __restrict__ y,          // (B,64,NK)
                                int C) {
    int lane = threadIdx.x;
    int half = lane >> 4, l15 = lane & 15;
    int col0 = blockIdx.x * 16;
    int o0   = blockIdx.y * 16;
    int b    = blockIdx.z;
    const __bf16* xb = xbf + (size_t)b * NPT * CMAXP;
    const int*    ib = idx + (size_t)b * NPT * KNB;

    int col = col0 + l15;
    int npt = col / KNB;
    int kk  = col - npt * KNB;
    int nbr = ib[npt * KNB + kk];
    const __bf16* ctr = xb + (size_t)npt * CMAXP;
    const __bf16* nrw = xb + (size_t)nbr * CMAXP;

    v8f acc = {};
    if (C == 64) {
        const __bf16* wrow = Wbf + (o0 + l15) * 128;
#pragma unroll
        for (int c0 = 0; c0 < 128; c0 += 32) {
            int klo = c0 + 8 * half;
            int khi = klo + 16;
            v16bf a = cat16(ldv8(wrow + klo), ldv8(wrow + khi));
            v16bf bm;
            if (c0 < 64) {
                v8bf nlo = ldv8(nrw + klo), clo = ldv8(ctr + klo);
                v8bf nhi = ldv8(nrw + khi), chi = ldv8(ctr + khi);
                v8bf dlo, dhi;
#pragma unroll
                for (int e = 0; e < 8; ++e) {
                    dlo[e] = (__bf16)((float)nlo[e] - (float)clo[e]);
                    dhi[e] = (__bf16)((float)nhi[e] - (float)chi[e]);
                }
                bm = cat16(dlo, dhi);
            } else {
                bm = cat16(ldv8(ctr + klo - 64), ldv8(ctr + khi - 64));
            }
            acc = __builtin_amdgcn_wmma_f32_16x16x32_bf16(
                false, a, false, bm, (short)0, acc, false, false);
        }
    } else {  // C == 3, twoC = 6 padded to 32; single K-step
        const __bf16* wrow = Wbf + (o0 + l15) * 32;
        v16bf a = {}, bm = {};
#pragma unroll
        for (int e = 0; e < 16; ++e) {
            int cp = wmma_kmap(e, half);
            a[e] = wrow[cp];                          // zero-padded weights
            // channels >= 3 of xbf rows are zero, so diff self-zeroes there
            float diff = (float)nrw[cp] - (float)ctr[cp];
            bool  isc  = (cp >= 3) && (cp < 6);
            float cen  = (float)ctr[isc ? (cp - 3) : 0];
            bm[e] = (__bf16)(diff + (isc ? cen : 0.0f));
        }
        acc = __builtin_amdgcn_wmma_f32_16x16x32_bf16(
            false, a, false, bm, (short)0, acc, false, false);
    }
    float* yb = y + ((size_t)b * 64 + o0) * NK;
#pragma unroll
    for (int j = 0; j < 8; ++j)
        yb[(size_t)(j + 8 * half) * NK + col] = acc[j];
}

// ---------------------------------------------------------------------------
// BN stats, deterministic two stages: per-block LDS tree -> partials,
// then fixed-order final sum. stats[c] = sum, stats[64+c] = sumsq.
// ---------------------------------------------------------------------------
__global__ void bn_stats_partial_kernel(const float* __restrict__ y,
                                        float* __restrict__ part) { // (64,32,2)
    int c = blockIdx.y;
    float s1 = 0.f, s2 = 0.f;
    for (int b = 0; b < NBATCH; ++b) {
        const float* p = y + ((size_t)b * 64 + c) * NK;
        for (int i = blockIdx.x * blockDim.x + threadIdx.x; i < NK;
             i += gridDim.x * blockDim.x) {
            float v = p[i]; s1 += v; s2 += v * v;
        }
    }
    __shared__ float sh1[256], sh2[256];
    sh1[threadIdx.x] = s1; sh2[threadIdx.x] = s2;
    __syncthreads();
    for (int off = 128; off > 0; off >>= 1) {
        if (threadIdx.x < off) {
            sh1[threadIdx.x] += sh1[threadIdx.x + off];
            sh2[threadIdx.x] += sh2[threadIdx.x + off];
        }
        __syncthreads();
    }
    if (threadIdx.x == 0) {
        part[(c * 32 + blockIdx.x) * 2 + 0] = sh1[0];
        part[(c * 32 + blockIdx.x) * 2 + 1] = sh2[0];
    }
}

__global__ void bn_stats_final_kernel(const float* __restrict__ part,
                                      float* __restrict__ stats) {
    int c = threadIdx.x;
    if (c >= 64) return;
    float s1 = 0.f, s2 = 0.f;
    for (int i = 0; i < 32; ++i) {
        s1 += part[(c * 32 + i) * 2 + 0];
        s2 += part[(c * 32 + i) * 2 + 1];
    }
    stats[c] = s1; stats[64 + c] = s2;
}

// Normalize + leaky-relu in place, plus max over k -> next-layer activation.
__global__ void bn_act_max_kernel(float* __restrict__ y,
                                  const float* __restrict__ stats,
                                  const float* __restrict__ gamma,
                                  const float* __restrict__ beta,
                                  float* __restrict__ xout) { // (B,64,N) or null
    int t = blockIdx.x * blockDim.x + threadIdx.x;
    if (t >= NBATCH * 64 * NPT) return;
    int n  = t % NPT;
    int bc = t / NPT;
    int c  = bc % 64;
    const float inv = 1.0f / (float)(NBATCH * NK);
    float mean = stats[c] * inv;
    float var  = stats[64 + c] * inv - mean * mean;
    float sc = gamma[c] * rsqrtf(var + 1e-5f);
    float sh = beta[c] - mean * sc;
    float* p = y + (size_t)bc * NK + (size_t)n * KNB;
    float mx = -3.0e38f;
#pragma unroll
    for (int j = 0; j < KNB; ++j) {
        float v = p[j] * sc + sh;
        v = (v > 0.f) ? v : 0.01f * v;
        p[j] = v;
        if (v > mx) mx = v;
    }
    if (xout) xout[(size_t)bc * NPT + n] = mx;
}

// ---------------------------------------------------------------------------
// Skip GEMM: skip += Wskip[:, 64L:64L+64] x act  (WMMA, C read from memory)
// A fetched as b128 chunks; B is coalesced b32 per element (f32 act).
// ---------------------------------------------------------------------------
__global__ void skip_gemm_kernel(const __bf16* __restrict__ Wsbf, // (64,256)
                                 const float*  __restrict__ act,  // (B,64,NK)
                                 float* __restrict__ skip,        // (B,64,NK)
                                 int cbase, int accumulate) {
    int lane = threadIdx.x;
    int half = lane >> 4, l15 = lane & 15;
    int col0 = blockIdx.x * 16;
    int o0   = blockIdx.y * 16;
    int b    = blockIdx.z;
    const float* ab = act  + (size_t)b * 64 * NK;
    float*       sb = skip + (size_t)b * 64 * NK;
    const __bf16* wrow = Wsbf + (o0 + l15) * 256 + cbase;
    v8f acc = {};
    if (accumulate) {
#pragma unroll
        for (int j = 0; j < 8; ++j)
            acc[j] = sb[(size_t)(o0 + j + 8 * half) * NK + col0 + l15];
    }
#pragma unroll
    for (int c0 = 0; c0 < 64; c0 += 32) {
        int klo = c0 + 8 * half;
        int khi = klo + 16;
        v16bf a = cat16(ldv8(wrow + klo), ldv8(wrow + khi));
        v16bf bm;
#pragma unroll
        for (int e = 0; e < 16; ++e) {
            int cp = c0 + wmma_kmap(e, half);
            bm[e] = (__bf16)ab[(size_t)cp * NK + col0 + l15];
        }
        acc = __builtin_amdgcn_wmma_f32_16x16x32_bf16(
            false, a, false, bm, (short)0, acc, false, false);
    }
#pragma unroll
    for (int j = 0; j < 8; ++j)
        sb[(size_t)(o0 + j + 8 * half) * NK + col0 + l15] = acc[j];
}

__global__ void final_max_kernel(const float* __restrict__ skip,
                                 float* __restrict__ out) {
    int t = blockIdx.x * blockDim.x + threadIdx.x;
    if (t >= NBATCH * 64 * NPT) return;
    const float* p = skip + (size_t)(t / NPT) * NK + (size_t)(t % NPT) * KNB;
    float mx = p[0];
#pragma unroll
    for (int j = 1; j < KNB; ++j) mx = (p[j] > mx) ? p[j] : mx;
    out[t] = mx;
}

// ---------------------------------------------------------------------------
// Host orchestration (graph-capture safe: launches only, all on `stream`).
// ---------------------------------------------------------------------------
extern "C" void kernel_launch(void* const* d_in, const int* in_sizes, int n_in,
                              void* d_out, int out_size, void* d_ws, size_t ws_size,
                              hipStream_t stream) {
    const float* x     = (const float*)d_in[0];
    const float* Wc[4] = {(const float*)d_in[1], (const float*)d_in[2],
                          (const float*)d_in[3], (const float*)d_in[4]};
    const float* Wskip = (const float*)d_in[5];
    const float* gam[4] = {(const float*)d_in[6],  (const float*)d_in[8],
                           (const float*)d_in[10], (const float*)d_in[12]};
    const float* bet[4] = {(const float*)d_in[7],  (const float*)d_in[9],
                           (const float*)d_in[11], (const float*)d_in[13]};

    // Bump allocation out of d_ws (256B aligned). Total ~192 MB.
    char* wp = (char*)d_ws;
    auto bump = [&](size_t bytes) -> void* {
        void* p = (void*)wp;
        wp += (bytes + 255) & ~(size_t)255;
        return p;
    };
    float*  xa    = (float*) bump((size_t)NBATCH * CMAXP * NPT * 4); // (B,64,N) f32
    __bf16* xbf   = (__bf16*)bump((size_t)NBATCH * NPT * CMAXP * 2); // (B,N,64) bf16
    float*  xx    = (float*) bump((size_t)NBATCH * NPT * 4);
    float*  pd    = (float*) bump((size_t)NPT * NPT * 4);
    int*    idxb  = (int*)   bump((size_t)NBATCH * NPT * KNB * 4);
    float*  ybuf  = (float*) bump((size_t)NBATCH * 64 * NK * 4);
    float*  skipb = (float*) bump((size_t)NBATCH * 64 * NK * 4);
    __bf16* wbf   = (__bf16*)bump((size_t)(64 * 32 + 3 * 64 * 128 + 64 * 256) * 2);
    float*  part  = (float*) bump((size_t)64 * 32 * 2 * 4);
    float*  stats = (float*) bump(128 * 4);

    // Padded bf16 weight arena: W1 (64x6->64x32), W2..4 (64x128), Wskip (64x256)
    const int woff[4]  = {0, 2048, 2048 + 8192, 2048 + 2 * 8192};
    const int wskipoff = 2048 + 3 * 8192;
    wcast_pad_kernel<<<(64 * 32 + 255) / 256, 256, 0, stream>>>(Wc[0], wbf + woff[0], 64, 6, 32);
    for (int i = 1; i < 4; ++i)
        wcast_pad_kernel<<<(64 * 128 + 255) / 256, 256, 0, stream>>>(Wc[i], wbf + woff[i], 64, 128, 128);
    wcast_pad_kernel<<<(64 * 256 + 255) / 256, 256, 0, stream>>>(Wskip, wbf + wskipoff, 64, 256, 256);

    transpose_in_kernel<<<(NBATCH * NPT + 255) / 256, 256, 0, stream>>>(x, xa);

    const int Cin[4] = {3, 64, 64, 64};

    for (int L = 0; L < 4; ++L) {
        int C = Cin[L];
        int Cpad = (C == 3) ? 32 : 64;
        prep_kernel<<<(NBATCH * NPT + 255) / 256, 256, 0, stream>>>(xa, xbf, xx, C);

        // knn per batch (pd slab reused; stream order serializes correctly)
        for (int b = 0; b < NBATCH; ++b) {
            dim3 gg(NPT / 16, NPT / 64);
            gram_pd_kernel<<<gg, 32, 0, stream>>>(
                xbf + (size_t)b * NPT * CMAXP, xx + (size_t)b * NPT, pd, Cpad);
            topk_kernel<<<NPT / 256, 256, 0, stream>>>(pd, idxb + (size_t)b * NPT * KNB);
        }

        dim3 ge(NK / 16, 4, NBATCH);
        edgeconv_kernel<<<ge, 32, 0, stream>>>(wbf + woff[L], xbf, idxb, ybuf, C);

        bn_stats_partial_kernel<<<dim3(32, 64), 256, 0, stream>>>(ybuf, part);
        bn_stats_final_kernel<<<1, 64, 0, stream>>>(part, stats);
        bn_act_max_kernel<<<(NBATCH * 64 * NPT + 255) / 256, 256, 0, stream>>>(
            ybuf, stats, gam[L], bet[L], (L < 3) ? xa : nullptr);

        skip_gemm_kernel<<<dim3(NK / 16, 4, NBATCH), 32, 0, stream>>>(
            wbf + wskipoff, ybuf, skipb, 64 * L, (L > 0) ? 1 : 0);
    }

    final_max_kernel<<<(NBATCH * 64 * NPT + 255) / 256, 256, 0, stream>>>(
        skipb, (float*)d_out);
}
